// GAT_85684597555228
// MI455X (gfx1250) — matrix operand order repaired and checked by reading, not compiled, not run on hardware
//
#include <hip/hip_runtime.h>

// ---------------------------------------------------------------------------
// 2-layer GAT for MI455X (gfx1250). GEMMs use v_wmma_f32_16x16x32_bf16
// (wave32, one wave per 16x16 output tile). Edge softmax + aggregation use
// L2-resident gather + global_atomic_add_f32 scatter (entire working set
// ~118 MB < 192 MB L2).
// ---------------------------------------------------------------------------

typedef __attribute__((ext_vector_type(16))) __bf16 v16bf;
typedef __attribute__((ext_vector_type(8)))  float  v8f;

union BF16x16 { v16bf v; __bf16 e[16]; };
union F32x8   { v8f   v; float  f[8];  };

__device__ __forceinline__ void cvt4(__bf16* d, float4 f) {
    d[0] = (__bf16)f.x; d[1] = (__bf16)f.y; d[2] = (__bf16)f.z; d[3] = (__bf16)f.w;
}

// ---------------- WMMA GEMM 1: h1[50000,256] = x[50000,128] @ W1[128,256] ---
__global__ void __launch_bounds__(256)
k_gemm1_wmma(const float* __restrict__ X, const float* __restrict__ W,
             float* __restrict__ H1)
{
    const int lane = threadIdx.x & 31;
    const int wid  = (blockIdx.x * 256 + threadIdx.x) >> 5;   // 0..49999
    const int mT   = wid >> 4;                                 // 0..3124
    const int nT   = wid & 15;                                 // 0..15
    const int row  = mT * 16 + (lane & 15);
    const int col  = nT * 16 + (lane & 15);
    const int kbA  = (lane >> 4) * 8;    // A: lane half picks K group +0 / +8
    const int kbB  = (lane >> 4) * 16;   // unused: B-frag indexes K with kbA, same order as A

    F32x8 acc = {};
    for (int k0 = 0; k0 < 128; k0 += 32) {
        BF16x16 aF, bF;
        // A-frag: e[0..7] = K k0+kbA..+7 ; e[8..15] = K k0+16+kbA..+7
        cvt4(&aF.e[0],  *(const float4*)(X + row * 128 + k0 + kbA));
        cvt4(&aF.e[4],  *(const float4*)(X + row * 128 + k0 + kbA + 4));
        cvt4(&aF.e[8],  *(const float4*)(X + row * 128 + k0 + 16 + kbA));
        cvt4(&aF.e[12], *(const float4*)(X + row * 128 + k0 + 16 + kbA + 4));
        // B-frag: e[j] = W1[k0+kbA+j (j<8) / k0+16+kbA+j-8 (j>=8), col]
        #pragma unroll
        for (int j = 0; j < 16; ++j)
            bF.e[j] = (__bf16)W[(k0 + (j < 8 ? kbA + j : 16 + kbA + j - 8)) * 256 + col];
        acc.v = __builtin_amdgcn_wmma_f32_16x16x32_bf16(
                    false, aF.v, false, bF.v, (short)0, acc.v, false, false);
    }
    const int mBase = mT * 16 + ((lane >> 4) << 3);
    #pragma unroll
    for (int v = 0; v < 8; ++v)
        H1[(mBase + v) * 256 + col] = acc.f[v];
}

// ---------------- WMMA GEMM 2: h2[200000,32] = out1[200000,64] @ W2[64,32] --
__global__ void __launch_bounds__(256)
k_gemm2_wmma(const float* __restrict__ A, const float* __restrict__ W,
             float* __restrict__ H2)
{
    const int lane = threadIdx.x & 31;
    const int wid  = (blockIdx.x * 256 + threadIdx.x) >> 5;   // 0..24999
    const int mT   = wid >> 1;                                 // 0..12499
    const int nT   = wid & 1;                                  // 0..1
    const int row  = mT * 16 + (lane & 15);
    const int col  = nT * 16 + (lane & 15);
    const int kbA  = (lane >> 4) * 8;
    const int kbB  = (lane >> 4) * 16;

    F32x8 acc = {};
    for (int k0 = 0; k0 < 64; k0 += 32) {
        BF16x16 aF, bF;
        cvt4(&aF.e[0],  *(const float4*)(A + row * 64 + k0 + kbA));
        cvt4(&aF.e[4],  *(const float4*)(A + row * 64 + k0 + kbA + 4));
        cvt4(&aF.e[8],  *(const float4*)(A + row * 64 + k0 + 16 + kbA));
        cvt4(&aF.e[12], *(const float4*)(A + row * 64 + k0 + 16 + kbA + 4));
        #pragma unroll
        for (int j = 0; j < 16; ++j)
            bF.e[j] = (__bf16)W[(k0 + (j < 8 ? kbA + j : 16 + kbA + j - 8)) * 32 + col];
        acc.v = __builtin_amdgcn_wmma_f32_16x16x32_bf16(
                    false, aF.v, false, bF.v, (short)0, acc.v, false, false);
    }
    const int mBase = mT * 16 + ((lane >> 4) << 3);
    #pragma unroll
    for (int v = 0; v < 8; ++v)
        H2[(mBase + v) * 32 + col] = acc.f[v];
}

// ---------------- helpers ---------------------------------------------------
__global__ void k_fill_u32(unsigned int* __restrict__ p, unsigned int v, int n) {
    for (int i = blockIdx.x * blockDim.x + threadIdx.x; i < n; i += gridDim.x * blockDim.x)
        p[i] = v;
}

// el/er: out[row*C + c] = sum_f h[row*C*F + c*F + f] * att[c*F + f]
__global__ void k_nodedot(const float* __restrict__ h,
                          const float* __restrict__ attL, const float* __restrict__ attR,
                          float* __restrict__ el, float* __restrict__ er,
                          int rows, int C, int F)
{
    int i = blockIdx.x * blockDim.x + threadIdx.x;
    if (i >= rows * C) return;
    int r = i / C, c = i % C;
    const float* hp = h + (size_t)r * C * F + c * F;
    const float* aL = attL + c * F;
    const float* aR = attR + c * F;
    float sl = 0.f, sr = 0.f;
    for (int f = 0; f < F; ++f) { float v = hp[f]; sl += v * aL[f]; sr += v * aR[f]; }
    el[i] = sl; er[i] = sr;
}

__device__ __forceinline__ float leaky(float v) {
    return fmaxf(v, 0.f) + 0.2f * fminf(v, 0.f);
}
__device__ __forceinline__ unsigned int f2key(float v) {
    unsigned int b = __float_as_uint(v);
    return (b & 0x80000000u) ? ~b : (b | 0x80000000u);
}
__device__ __forceinline__ float key2f(unsigned int k) {
    return (k & 0x80000000u) ? __uint_as_float(k & 0x7fffffffu) : __uint_as_float(~k);
}

__global__ void k_edge_max(const int* __restrict__ src, const int* __restrict__ dst,
                           const float* __restrict__ el, const float* __restrict__ er,
                           unsigned int* __restrict__ mkey, int E, int C)
{
    int i = blockIdx.x * blockDim.x + threadIdx.x;
    if (i >= E * C) return;
    int e = i / C, c = i % C;
    float v = leaky(el[src[e] * C + c] + er[dst[e] * C + c]);
    atomicMax(&mkey[dst[e] * C + c], f2key(v));
}

__global__ void k_edge_exp(const int* __restrict__ src, const int* __restrict__ dst,
                           const float* __restrict__ el, const float* __restrict__ er,
                           const unsigned int* __restrict__ mkey,
                           float* __restrict__ den, float* __restrict__ abuf, int E, int C)
{
    int i = blockIdx.x * blockDim.x + threadIdx.x;
    if (i >= E * C) return;
    int e = i / C, c = i % C;
    int d = dst[e] * C + c;
    float v = leaky(el[src[e] * C + c] + er[d]);
    float a = __expf(v - key2f(mkey[d]));
    abuf[i] = a;
    atomicAdd(&den[d], a);
}

// num[dst, c, f] += a[e,c] * h[src, c, f]; stride = C*F, F4 = F/4
__global__ void k_edge_scatter(const int* __restrict__ src, const int* __restrict__ dst,
                               const float* __restrict__ abuf, const float* __restrict__ h,
                               float* __restrict__ num, int E, int C, int F4, int stride)
{
    int i = blockIdx.x * blockDim.x + threadIdx.x;
    int per = C * F4;
    if (i >= E * per) return;
    int e = i / per, r = i % per, c = r / F4, f4 = r % F4;
    float a = abuf[e * C + c];
    const float4 hv = *(const float4*)(h + (size_t)src[e] * stride + c * (F4 * 4) + f4 * 4);
    float* o = num + (size_t)dst[e] * stride + c * (F4 * 4) + f4 * 4;
    atomicAdd(o + 0, a * hv.x);
    atomicAdd(o + 1, a * hv.y);
    atomicAdd(o + 2, a * hv.z);
    atomicAdd(o + 3, a * hv.w);
}

// out1 = relu(num/den + b1), in place into num. [N,4,64]
__global__ void k_final1(float* __restrict__ num, const float* __restrict__ den,
                         const float* __restrict__ b1, int N)
{
    int i = blockIdx.x * blockDim.x + threadIdx.x;
    if (i >= N * 256) return;
    int n = i / 256, r = i % 256, c = r / 64;
    float d = den[n * 4 + c];
    float v = (d > 0.f) ? num[i] / d : 0.f;
    num[i] = fmaxf(v + b1[r], 0.f);
}

// out[n, h2, cls] = mean_hl( num2[n,hl,h2,cls]/den2[n,hl,h2] + b2[h2,cls] )
__global__ void k_final2(const float* __restrict__ num, const float* __restrict__ den,
                         const float* __restrict__ b2, float* __restrict__ out, int N)
{
    int i = blockIdx.x * blockDim.x + threadIdx.x;
    if (i >= N * 32) return;
    int n = i / 32, r = i % 32, h2 = r >> 3;
    float acc = 0.f;
    #pragma unroll
    for (int hl = 0; hl < 4; ++hl) {
        float d = den[n * 16 + hl * 4 + h2];
        float v = (d > 0.f) ? num[n * 128 + hl * 32 + r] / d : 0.f;
        acc += v + b2[r];
    }
    out[i] = acc * 0.25f;
}

// ---------------------------------------------------------------------------
extern "C" void kernel_launch(void* const* d_in, const int* in_sizes, int n_in,
                              void* d_out, int out_size, void* d_ws, size_t ws_size,
                              hipStream_t stream)
{
    const float* x   = (const float*)d_in[0];
    const float* W1  = (const float*)d_in[1];
    const float* al1 = (const float*)d_in[2];
    const float* ar1 = (const float*)d_in[3];
    const float* b1  = (const float*)d_in[4];
    const float* W2  = (const float*)d_in[5];
    const float* al2 = (const float*)d_in[6];
    const float* ar2 = (const float*)d_in[7];
    const float* b2  = (const float*)d_in[8];
    const int*   src = (const int*)d_in[9];
    const int*   dst = (const int*)d_in[10];

    const int N = in_sizes[0] / 128;   // 50000 (divisible by 16)
    const int E = in_sizes[9];         // 800000

    float* ws = (float*)d_ws;
    // workspace layout (float offsets), regions reused across layers; peak 118.4 MB
    const size_t O_H1   = 0;                       // N*256
    const size_t O_NUM1 = (size_t)N * 256;         // N*256 (becomes out1)
    const size_t O_EL1  = O_NUM1 + (size_t)N * 256;
    const size_t O_ER1  = O_EL1 + (size_t)N * 4;
    const size_t O_M1   = O_ER1 + (size_t)N * 4;
    const size_t O_DEN1 = O_M1  + (size_t)N * 4;
    const size_t O_A1   = O_DEN1 + (size_t)N * 4;  // E*4
    // layer 2 (reuses dead layer-1 regions)
    const size_t O_H2   = 0;                       // N*128
    const size_t O_NUM2 = (size_t)N * 128;         // N*128 (inside dead H1)
    const size_t O_EL2  = (size_t)N * 256;         // reuses out1 region after gemm2
    const size_t O_ER2  = O_EL2 + (size_t)N * 16;
    const size_t O_M2   = O_ER2 + (size_t)N * 16;
    const size_t O_DEN2 = O_M2  + (size_t)N * 16;
    const size_t O_A2   = O_DEN2 + (size_t)N * 16; // E*16

    const int T = 256;
    auto blocks = [](long total, int t) { return (int)((total + t - 1) / t); };

    // ---- layer 1 init (fresh every call: graph replay determinism) ----
    k_fill_u32<<<blocks((long)N * 256, T), T, 0, stream>>>((unsigned int*)(ws + O_NUM1), 0u, N * 256);
    k_fill_u32<<<blocks((long)N * 4, T), T, 0, stream>>>((unsigned int*)(ws + O_DEN1), 0u, N * 4);
    k_fill_u32<<<blocks((long)N * 4, T), T, 0, stream>>>((unsigned int*)(ws + O_M1), 0u, N * 4);

    // ---- layer 1 ----
    k_gemm1_wmma<<<(N / 16) * 16 / 8, T, 0, stream>>>(x, W1, ws + O_H1);
    k_nodedot<<<blocks((long)N * 4, T), T, 0, stream>>>(ws + O_H1, al1, ar1,
                                                        ws + O_EL1, ws + O_ER1, N, 4, 64);
    k_edge_max<<<blocks((long)E * 4, T), T, 0, stream>>>(src, dst, ws + O_EL1, ws + O_ER1,
                                                         (unsigned int*)(ws + O_M1), E, 4);
    k_edge_exp<<<blocks((long)E * 4, T), T, 0, stream>>>(src, dst, ws + O_EL1, ws + O_ER1,
                                                         (const unsigned int*)(ws + O_M1),
                                                         ws + O_DEN1, ws + O_A1, E, 4);
    k_edge_scatter<<<blocks((long)E * 4 * 16, T), T, 0, stream>>>(src, dst, ws + O_A1,
                                                                  ws + O_H1, ws + O_NUM1,
                                                                  E, 4, 16, 256);
    k_final1<<<blocks((long)N * 256, T), T, 0, stream>>>(ws + O_NUM1, ws + O_DEN1, b1, N);

    // ---- layer 2 init ----
    k_fill_u32<<<blocks((long)N * 128, T), T, 0, stream>>>((unsigned int*)(ws + O_NUM2), 0u, N * 128);
    k_fill_u32<<<blocks((long)N * 16, T), T, 0, stream>>>((unsigned int*)(ws + O_DEN2), 0u, N * 16);
    k_fill_u32<<<blocks((long)N * 16, T), T, 0, stream>>>((unsigned int*)(ws + O_M2), 0u, N * 16);

    // ---- layer 2 ----
    k_gemm2_wmma<<<((N * 4) / 16) * 2 / 8, T, 0, stream>>>(ws + O_NUM1, W2, ws + O_H2);
    k_nodedot<<<blocks((long)N * 16, T), T, 0, stream>>>(ws + O_H2, al2, ar2,
                                                         ws + O_EL2, ws + O_ER2, N * 4, 4, 8);
    k_edge_max<<<blocks((long)E * 16, T), T, 0, stream>>>(src, dst, ws + O_EL2, ws + O_ER2,
                                                          (unsigned int*)(ws + O_M2), E, 16);
    k_edge_exp<<<blocks((long)E * 16, T), T, 0, stream>>>(src, dst, ws + O_EL2, ws + O_ER2,
                                                          (const unsigned int*)(ws + O_M2),
                                                          ws + O_DEN2, ws + O_A2, E, 16);
    k_edge_scatter<<<blocks((long)E * 16 * 2, T), T, 0, stream>>>(src, dst, ws + O_A2,
                                                                  ws + O_H2, ws + O_NUM2,
                                                                  E, 16, 2, 128);
    k_final2<<<blocks((long)N * 32, T), T, 0, stream>>>(ws + O_NUM2, ws + O_DEN2, b2,
                                                        (float*)d_out, N);
    (void)n_in; (void)out_size; (void)ws_size;
}
